// DiscreteTimeS4_39616778338839
// MI455X (gfx1250) — compile-verified, matrix operand order checked by reading, not run on hardware
//
#include <hip/hip_runtime.h>

// Problem dims
#define BATCH 64
#define T     2048
#define INF   64
#define P     128
#define HD    256
#define OUTF  64
#define RROWS (BATCH*T)   // 131072
#define NCH   8
#define CLEN  (T/NCH)     // 256

typedef __attribute__((ext_vector_type(16))) __bf16 v16bf;
typedef __attribute__((ext_vector_type(8)))  float  v8f;

__device__ __forceinline__ v8f wmma_bf16(v16bf a, v16bf b, v8f c) {
  return __builtin_amdgcn_wmma_f32_16x16x32_bf16(
      /*neg_a=*/false, a, /*neg_b=*/false, b,
      /*c_mod=*/(short)0, c, /*reuse_a=*/false, /*reuse_b=*/false);
}

// K index inside a 16-bit A/B fragment (ISA 7.12.2): VGPR v holds 2 elements,
// v<4 -> K=2v..2v+1 (+8 for lanes16-31), v>=4 -> K=16+2(v-4).. (+8 for hi lanes)
__device__ __forceinline__ int frag_k(int e, int half) {
  int v = e >> 1, w = e & 1;
  return (v < 4 ? 2 * v : 2 * (v - 4) + 16) + half * 8 + w;
}

// A fragment from bf16 row-major source (two contiguous 16B runs per lane)
__device__ __forceinline__ v16bf load_a(const __bf16* s, int ld) {
  int lane = threadIdx.x & 31;
  int half = lane >> 4, m = lane & 15;
  v16bf f;
#pragma unroll
  for (int e = 0; e < 16; ++e) f[e] = s[m * ld + frag_k(e, half)];
  return f;
}
// A fragment gathered from f32 source with on-the-fly bf16 convert
__device__ __forceinline__ v16bf load_a_f32(const float* s, int ld) {
  int lane = threadIdx.x & 31;
  int half = lane >> 4, m = lane & 15;
  v16bf f;
#pragma unroll
  for (int e = 0; e < 16; ++e) f[e] = (__bf16)s[m * ld + frag_k(e, half)];
  return f;
}
// B fragment from K-major source: s[k*ld + n]  (B is K x N)
__device__ __forceinline__ v16bf load_b(const __bf16* s, int ld) {
  int lane = threadIdx.x & 31;
  int half = lane >> 4, n = lane & 15;
  v16bf f;
#pragma unroll
  for (int e = 0; e < 16; ++e) f[e] = s[frag_k(e, half) * ld + n];
  return f;
}
// B fragment from N-major (transposed) source: s[n*ld + k]
__device__ __forceinline__ v16bf load_bt(const __bf16* s, int ld) {
  int lane = threadIdx.x & 31;
  int half = lane >> 4, n = lane & 15;
  v16bf f;
#pragma unroll
  for (int e = 0; e < 16; ++e) f[e] = s[n * ld + frag_k(e, half)];
  return f;
}

// Async global->LDS 16B copy (ASYNCcnt-tracked); loff/goff are byte offsets.
__device__ __forceinline__ void async_b128(unsigned loff, unsigned goff,
                                           const void* sbase) {
  asm volatile("global_load_async_to_lds_b128 %0, %1, %2 offset:0"
               :: "v"(loff), "v"(goff), "s"(sbase)
               : "memory");
}
__device__ __forceinline__ void wait_async0() {
  asm volatile("s_wait_asynccnt 0" ::: "memory");
}

// ---------------- weight f32 -> bf16 ----------------
__global__ void k_cvt(const float* __restrict__ s, __bf16* __restrict__ d, int n) {
  int i = blockIdx.x * 256 + threadIdx.x;
  if (i < n) d[i] = (__bf16)s[i];
}

// ---------------- phase 1: U = relu(x @ Win^T + b_in) @ B ----------------
// 64 rows per block (4 waves). x tile staged as raw f32 via async-to-LDS.
#define XS_LDF (INF + 4)   // f32 row stride 272B: 16B aligned, 17-bank skew
#define XP_LD  (P + 8)     // bf16 row stride 272B
__global__ __launch_bounds__(128) void k_in_proj(
    const float* __restrict__ x, const float* __restrict__ b_in,
    const __bf16* __restrict__ Winb, const __bf16* __restrict__ Bb,
    __bf16* __restrict__ U) {
  __shared__ __align__(16) float  xs[64 * XS_LDF];   // 17 KB
  __shared__ __align__(16) __bf16 xp[64 * XP_LD];    // 17 KB
  int tid = threadIdx.x;
  int row0 = blockIdx.x * 64;
  // async-stage 64x64 f32 tile: 1024 x b128, 8 per thread
  {
    const void* src = (const void*)(x + (size_t)row0 * INF);
    unsigned lbase = (unsigned)(uintptr_t)(&xs[0]);
#pragma unroll
    for (int i = 0; i < 8; ++i) {
      int fl = i * 128 + tid;
      int r = fl >> 4, j = fl & 15;            // 16 x 16B per row
      async_b128(lbase + (unsigned)(r * (XS_LDF * 4) + j * 16),
                 (unsigned)(fl * 16), src);
    }
    wait_async0();
  }
  __syncthreads();

  int w = tid >> 5, lane = tid & 31;
  int col = lane & 15, rbase = (lane >> 4) * 8;
  const float* xsw = xs + (w * 16) * XS_LDF;
  v16bf a0 = load_a_f32(xsw, XS_LDF);
  v16bf a1 = load_a_f32(xsw + 32, XS_LDF);
  __bf16* xpw = xp + (w * 16) * XP_LD;

  // GEMM1: x_proj[16 x 128] = x[16x64] @ Win^T, + bias, relu
#pragma unroll
  for (int p0 = 0; p0 < P / 16; ++p0) {
    v8f acc = {};
    acc = wmma_bf16(a0, load_bt(Winb + (p0 * 16) * INF, INF), acc);
    acc = wmma_bf16(a1, load_bt(Winb + (p0 * 16) * INF + 32, INF), acc);
    float bias = b_in[p0 * 16 + col];
#pragma unroll
    for (int r = 0; r < 8; ++r) {
      float v = acc[r] + bias;
      v = v > 0.f ? v : 0.f;
      xpw[(rbase + r) * XP_LD + p0 * 16 + col] = (__bf16)v;
    }
  }
  // GEMM2: U[16 x 256] = x_proj @ B   (same-wave LDS RAW, DScnt-tracked)
  v16bf pa[4];
#pragma unroll
  for (int kk = 0; kk < 4; ++kk) pa[kk] = load_a(xpw + kk * 32, XP_LD);
#pragma unroll
  for (int h0 = 0; h0 < HD / 16; ++h0) {
    v8f acc = {};
#pragma unroll
    for (int kk = 0; kk < 4; ++kk)
      acc = wmma_bf16(pa[kk], load_b(Bb + (kk * 32) * HD + h0 * 16, HD), acc);
#pragma unroll
    for (int r = 0; r < 8; ++r)
      U[(size_t)(row0 + w * 16 + rbase + r) * HD + h0 * 16 + col] = (__bf16)acc[r];
  }
}

// ---------------- phase 2: chunked diagonal scan ----------------
// 2a: per-chunk partial state (zero init). block = (b, chunk), thread = h.
__global__ __launch_bounds__(HD) void k_scan_partial(
    const float* __restrict__ a, const __bf16* __restrict__ U,
    float* __restrict__ Spart) {
  int h = threadIdx.x;
  int bc = blockIdx.x;                 // b*NCH + c
  int b = bc / NCH, c = bc % NCH;
  float af = a[h];
  size_t base = ((size_t)b * T + (size_t)c * CLEN) * HD + h;
  float s = 0.f;
#pragma unroll 8
  for (int t = 0; t < CLEN; ++t)
    s = fmaf(af, s, (float)U[base + (size_t)t * HD]);
  Spart[(size_t)bc * HD + h] = s;
}
// 2b: combine chunk states with decay a^CLEN -> chunk initial states
__global__ __launch_bounds__(256) void k_scan_combine(
    const float* __restrict__ a, const float* __restrict__ Spart,
    float* __restrict__ Hin) {
  int idx = blockIdx.x * 256 + threadIdx.x;   // b*HD + h
  int h = idx & (HD - 1), b = idx / HD;
  float af = a[h];
  float aL = af;
#pragma unroll
  for (int i = 0; i < 8; ++i) aL = aL * aL;   // a^256
  float run = 0.f;
#pragma unroll
  for (int c = 0; c < NCH; ++c) {
    size_t o = ((size_t)b * NCH + c) * HD + h;
    Hin[o] = run;
    run = fmaf(aL, run, Spart[o]);
  }
}
// 2c: re-apply with correct initial state, write h_t in place over U (bf16)
__global__ __launch_bounds__(HD) void k_scan_apply(
    const float* __restrict__ a, const float* __restrict__ Hin, __bf16* U) {
  int h = threadIdx.x;
  int bc = blockIdx.x;
  int b = bc / NCH, c = bc % NCH;
  float af = a[h];
  float s = Hin[(size_t)bc * HD + h];
  size_t base = ((size_t)b * T + (size_t)c * CLEN) * HD + h;
#pragma unroll 8
  for (int t = 0; t < CLEN; ++t) {
    size_t o = base + (size_t)t * HD;
    s = fmaf(af, s, (float)U[o]);
    U[o] = (__bf16)s;
  }
}

// ---------------- phase 3: out = (Hs @ C) @ Wout^T + b_out ----------------
#define HS_LD (HD + 8)    // 528B rows: 16B aligned, 33-bank skew
__global__ __launch_bounds__(128) void k_out_proj(
    const __bf16* __restrict__ Hs, const __bf16* __restrict__ Cb,
    const __bf16* __restrict__ Woutb, const float* __restrict__ b_out,
    float* __restrict__ out) {
  __shared__ __align__(16) __bf16 hs[64 * HS_LD];   // 33 KB
  __shared__ __align__(16) __bf16 yp[64 * XP_LD];   // 17 KB
  int tid = threadIdx.x;
  int row0 = blockIdx.x * 64;
  // async-stage 64x256 bf16 tile: 2048 x b128, 16 per thread
  {
    const void* src = (const void*)(Hs + (size_t)row0 * HD);
    unsigned lbase = (unsigned)(uintptr_t)(&hs[0]);
#pragma unroll
    for (int i = 0; i < 16; ++i) {
      int fl = i * 128 + tid;
      int r = fl >> 5, j = fl & 31;            // 32 x 16B per row
      async_b128(lbase + (unsigned)(r * (HS_LD * 2) + j * 16),
                 (unsigned)(fl * 16), src);
    }
    wait_async0();
  }
  __syncthreads();

  int w = tid >> 5, lane = tid & 31;
  int col = lane & 15, rbase = (lane >> 4) * 8;
  const __bf16* hsw = hs + (w * 16) * HS_LD;
  v16bf ha[8];
#pragma unroll
  for (int kk = 0; kk < 8; ++kk) ha[kk] = load_a(hsw + kk * 32, HS_LD);
  __bf16* ypw = yp + (w * 16) * XP_LD;

  // y[16 x 128] = h @ C
#pragma unroll
  for (int p0 = 0; p0 < P / 16; ++p0) {
    v8f acc = {};
#pragma unroll
    for (int kk = 0; kk < 8; ++kk)
      acc = wmma_bf16(ha[kk], load_b(Cb + (kk * 32) * P + p0 * 16, P), acc);
#pragma unroll
    for (int r = 0; r < 8; ++r)
      ypw[(rbase + r) * XP_LD + p0 * 16 + col] = (__bf16)acc[r];
  }
  // out[16 x 64] = y @ Wout^T + b_out
  v16bf ya[4];
#pragma unroll
  for (int kk = 0; kk < 4; ++kk) ya[kk] = load_a(ypw + kk * 32, XP_LD);
#pragma unroll
  for (int o0 = 0; o0 < OUTF / 16; ++o0) {
    v8f acc = {};
#pragma unroll
    for (int kk = 0; kk < 4; ++kk)
      acc = wmma_bf16(ya[kk], load_bt(Woutb + (o0 * 16) * P + kk * 32, P), acc);
    float bias = b_out[o0 * 16 + col];
#pragma unroll
    for (int r = 0; r < 8; ++r)
      out[(size_t)(row0 + w * 16 + rbase + r) * OUTF + o0 * 16 + col] = acc[r] + bias;
  }
}

extern "C" void kernel_launch(void* const* d_in, const int* in_sizes, int n_in,
                              void* d_out, int out_size, void* d_ws, size_t ws_size,
                              hipStream_t stream) {
  const float* x    = (const float*)d_in[0];   // [64,2048,64]
  const float* a    = (const float*)d_in[1];   // [256]
  const float* Bm   = (const float*)d_in[2];   // [128,256]
  const float* Cm   = (const float*)d_in[3];   // [256,128]
  const float* Win  = (const float*)d_in[4];   // [128,64]
  const float* bin  = (const float*)d_in[5];   // [128]
  const float* Wout = (const float*)d_in[6];   // [64,128]
  const float* bout = (const float*)d_in[7];   // [64]
  float* out = (float*)d_out;

  char* ws = (char*)d_ws;
  __bf16* U = (__bf16*)ws;                       // [B*T, H] bf16, 64 MB (becomes Hs in place)
  size_t off = (size_t)RROWS * HD * sizeof(__bf16);
  __bf16* Winb  = (__bf16*)(ws + off); off += (size_t)P * INF * 2;
  __bf16* Bb    = (__bf16*)(ws + off); off += (size_t)P * HD * 2;
  __bf16* Cb    = (__bf16*)(ws + off); off += (size_t)HD * P * 2;
  __bf16* Woutb = (__bf16*)(ws + off); off += (size_t)OUTF * P * 2;
  float* Spart  = (float*)(ws + off);  off += (size_t)BATCH * NCH * HD * 4;
  float* Hin    = (float*)(ws + off);

  k_cvt<<<(P * INF + 255) / 256, 256, 0, stream>>>(Win, Winb, P * INF);
  k_cvt<<<(P * HD + 255) / 256, 256, 0, stream>>>(Bm, Bb, P * HD);
  k_cvt<<<(HD * P + 255) / 256, 256, 0, stream>>>(Cm, Cb, HD * P);
  k_cvt<<<(OUTF * P + 255) / 256, 256, 0, stream>>>(Wout, Woutb, OUTF * P);

  k_in_proj<<<RROWS / 64, 128, 0, stream>>>(x, bin, Winb, Bb, U);

  k_scan_partial<<<BATCH * NCH, HD, 0, stream>>>(a, U, Spart);
  k_scan_combine<<<(BATCH * HD) / 256, 256, 0, stream>>>(a, Spart, Hin);
  k_scan_apply<<<BATCH * NCH, HD, 0, stream>>>(a, Hin, U);

  k_out_proj<<<RROWS / 64, 128, 0, stream>>>(U, Cb, Woutb, bout, out);
}
